// TransformerEncoderLayer_87926570484370
// MI455X (gfx1250) — matrix operand order repaired and checked by reading, not compiled, not run on hardware
//
#include <hip/hip_runtime.h>
#include <math.h>
#include <stdint.h>

typedef _Float16 f16;
typedef __attribute__((ext_vector_type(16))) _Float16 v16h;
typedef __attribute__((ext_vector_type(8)))  float    v8f;
typedef __attribute__((ext_vector_type(4)))  unsigned int u32x4;
typedef __attribute__((ext_vector_type(4)))  int i32x4_t;

#define LDT 40  // LDS tile row stride in f16 elems (32 data + 8 pad -> 80B rows, 16B aligned)

// ---- CDNA5 async global->LDS copy (ASYNCcnt path), guarded with fallback ----------
#if defined(__has_builtin)
#if __has_builtin(__builtin_amdgcn_global_load_async_to_lds_b128)
#define HAVE_ASYNC_COPY 1
#endif
#if __has_builtin(__builtin_amdgcn_s_wait_asynccnt)
#define HAVE_WAIT_ASYNC 1
#endif
#endif

__device__ __forceinline__ void wait_tile_loads() {
#if defined(HAVE_ASYNC_COPY)
#if defined(HAVE_WAIT_ASYNC)
  __builtin_amdgcn_s_wait_asynccnt(0);
#else
  asm volatile("s_wait_asynccnt 0x0" ::: "memory");
#endif
#endif
}

// Stage a 128x32 f16 tile (global row stride ld) into LDS [row*LDT+k]; 256 threads.
// Async path: per-lane GLOBAL_LOAD_ASYNC_TO_LDS_B128 (no VGPR staging).
__device__ __forceinline__ void issue_tile(f16* dst, const f16* src, size_t ld, int t) {
#pragma unroll
  for (int i = 0; i < 2; ++i) {
    int idx = i * 256 + t;          // 512 chunks of 8 halfs
    int row = idx >> 2;
    int kp  = (idx & 3) << 3;
    const f16* g = src + (size_t)row * ld + kp;
    f16*       l = dst + row * LDT + kp;
#if defined(HAVE_ASYNC_COPY)
    __builtin_amdgcn_global_load_async_to_lds_b128(
        (__attribute__((address_space(1))) i32x4_t*)(uintptr_t)g,
        (__attribute__((address_space(3))) i32x4_t*)(unsigned int)(uintptr_t)l,
        0, 0);
#else
    *(u32x4*)l = *(const u32x4*)g;
#endif
  }
}

// ---------------- WMMA fragment loaders (CDNA5 16-bit layouts, wave32) -------------
// A 16x32: lane m=lane&15, half=lane>>4. VGPR v<4: K = half*8 + 2v,2v+1 ; v>=4: +16.
__device__ __forceinline__ v16h frag_A(const f16* As, int rowBase, int half) {
  const f16* p = As + rowBase;
  union { u32x4 q[2]; v16h v; } u;
  u.q[0] = *(const u32x4*)(p + half * 8);        // K = half*8 .. +7
  u.q[1] = *(const u32x4*)(p + 16 + half * 8);   // K = 16+half*8 .. +7
  return u.v;
}
// B 32x16 (stored B-friendly [n][k] in LDS): lane n=lane&15, half=lane>>4.
__device__ __forceinline__ v16h frag_B(const f16* Bs, int rowBase, int half) {
  const f16* p = Bs + rowBase + half * 8;
  union { u32x4 q[2]; v16h v; } u;
  u.q[0] = *(const u32x4*)(p);
  u.q[1] = *(const u32x4*)(p + 16);
  return u.v;
}

// ---------------- small prep kernels -----------------------------------------------
__global__ void init_scal_kernel(unsigned int* scal) {
  if (threadIdx.x == 0) { scal[0] = 0u; scal[1] = 0u; }
}

__global__ void centers_kernel(const float* __restrict__ bbox, float* cx, float* cy, int N) {
  int i = blockIdx.x * 256 + threadIdx.x;
  if (i < N) {
    cx[i] = (bbox[4 * i + 0] + bbox[4 * i + 2]) * 0.5f;
    cy[i] = (bbox[4 * i + 1] + bbox[4 * i + 3]) * 0.5f;
  }
}

__global__ void cast_f16_kernel(const float* __restrict__ in, f16* __restrict__ out, size_t n) {
  for (size_t i = (size_t)blockIdx.x * blockDim.x + threadIdx.x; i < n;
       i += (size_t)gridDim.x * blockDim.x)
    out[i] = (f16)in[i];
}

// out[c][r] = (f16) in[r][c];  in: [R][C] f32
__global__ __launch_bounds__(256) void transpose_cast_kernel(const float* __restrict__ in,
                                                             f16* __restrict__ out, int R, int C) {
  __shared__ float tile[32][33];
  int tx = threadIdx.x & 31, ty = threadIdx.x >> 5;
  int c0 = blockIdx.x * 32, r0 = blockIdx.y * 32;
#pragma unroll
  for (int j = 0; j < 32; j += 8) tile[ty + j][tx] = in[(size_t)(r0 + ty + j) * C + c0 + tx];
  __syncthreads();
#pragma unroll
  for (int j = 0; j < 32; j += 8)
    out[(size_t)(c0 + ty + j) * R + r0 + tx] = (f16)tile[tx][ty + j];
}

// mode 0: max over all pairs of sqrt(dx^2+dy^2+1e-6)
// mode 1: max over pairs (i!=j) of (|sgn dx|+|sgn dy|)*exp(-dist_norm*3), dist_norm uses scal[0]
__global__ __launch_bounds__(256) void pair_max_kernel(const float* __restrict__ cx,
                                                       const float* __restrict__ cy,
                                                       const float* __restrict__ scal,
                                                       unsigned int* outm, int N, int mode) {
  size_t total = (size_t)N * (size_t)N;
  float dinv = (mode == 1) ? (3.0f / (scal[0] + 1e-6f)) : 0.0f;
  float local = 0.0f;
  size_t stride = (size_t)gridDim.x * blockDim.x;
  for (size_t p = (size_t)blockIdx.x * blockDim.x + threadIdx.x; p < total; p += stride) {
    int i = (int)(p / (size_t)N), j = (int)(p % (size_t)N);
    float dx = cx[i] - cx[j], dy = cy[i] - cy[j];
    float d = sqrtf(dx * dx + dy * dy + 1e-6f);
    float v;
    if (mode == 0) v = d;
    else {
      v = ((dx != 0.0f ? 1.0f : 0.0f) + (dy != 0.0f ? 1.0f : 0.0f)) * __expf(-d * dinv);
      if (i == j) v = 0.0f;
    }
    local = fmaxf(local, v);
  }
#pragma unroll
  for (int m = 1; m < 32; m <<= 1) local = fmaxf(local, __shfl_xor(local, m, 32));
  if ((threadIdx.x & 31) == 0) atomicMax(outm, __float_as_uint(local));
}

// ---------------- attention scores: P' = exp(qk*scale + bias - 8) as fp16 ----------
__global__ __launch_bounds__(256) void attn_scores_kernel(
    const f16* __restrict__ S16, const float* __restrict__ cx, const float* __restrict__ cy,
    const float* __restrict__ scal, f16* __restrict__ P, int N, int D, float scale) {
  __shared__ f16 As[2][128 * LDT];
  __shared__ f16 Bs[2][128 * LDT];
  __shared__ float qx[128], qy[128], kx2[128], ky2[128];
  int t = threadIdx.x, lane = t & 31, wid = t >> 5;
  int half = lane >> 4, nl = lane & 15;
  int wm = wid >> 2, wn = wid & 3;
  int q0 = blockIdx.y * 128, c0 = blockIdx.x * 128;
  if (t < 128) { qx[t] = cx[q0 + t]; qy[t] = cy[q0 + t]; }
  else { int u = t - 128; kx2[u] = cx[c0 + u]; ky2[u] = cy[c0 + u]; }
  float dinv = 3.0f / (scal[0] + 1e-6f);
  float binv = 5.0f / (scal[1] + 1e-6f);

  v8f acc[4][2] = {};
  const f16* Ab = S16 + (size_t)q0 * D;
  const f16* Bb = S16 + (size_t)c0 * D;
  int nk = D / 32;
  issue_tile(As[0], Ab, D, t);
  issue_tile(Bs[0], Bb, D, t);
  for (int kt = 0; kt < nk; ++kt) {
    int cur = kt & 1;
    wait_tile_loads();     // this wave's tile-kt async loads landed in LDS
    __syncthreads();       // everyone's landed; everyone done reading spare buffer
    if (kt + 1 < nk) {     // prefetch next tile into the spare buffer
      issue_tile(As[cur ^ 1], Ab + (kt + 1) * 32, D, t);
      issue_tile(Bs[cur ^ 1], Bb + (kt + 1) * 32, D, t);
    }
    v16h af[4], bfr[2];
#pragma unroll
    for (int i = 0; i < 4; ++i) af[i] = frag_A(As[cur], (wm * 64 + i * 16 + nl) * LDT, half);
#pragma unroll
    for (int j = 0; j < 2; ++j) bfr[j] = frag_B(Bs[cur], (wn * 32 + j * 16 + nl) * LDT, half);
#pragma unroll
    for (int i = 0; i < 4; ++i)
#pragma unroll
      for (int j = 0; j < 2; ++j)
        acc[i][j] = __builtin_amdgcn_wmma_f32_16x16x32_f16(false, af[i], false, bfr[j],
                                                           (short)0, acc[i][j], false, false);
  }
#pragma unroll
  for (int i = 0; i < 4; ++i)
#pragma unroll
    for (int j = 0; j < 2; ++j)
#pragma unroll
      for (int r = 0; r < 8; ++r) {
        int row_l = wm * 64 + i * 16 + half * 8 + r;
        int col_l = wn * 32 + j * 16 + nl;
        int grow = q0 + row_l, gcol = c0 + col_l;
        float dx = qx[row_l] - kx2[col_l];
        float dy = qy[row_l] - ky2[col_l];
        float dist = sqrtf(dx * dx + dy * dy + 1e-6f);
        float bias = ((dx != 0.0f ? 1.0f : 0.0f) + (dy != 0.0f ? 1.0f : 0.0f)) * __expf(-dist * dinv);
        if (grow == gcol) bias = 0.0f;
        float s = acc[i][j][r] * scale + bias * binv;
        P[(size_t)grow * N + gcol] = (f16)__expf(s - 8.0f);  // constant shift cancels in softmax
      }
}

// ---------------- generic f16 WMMA GEMM, B-friendly [N][K] operand -----------------
// EPI 0: out_f32 = (A@B)[m][n] / rowsum(A[m][:])   (softmax normalize, deterministic)
// EPI 1: out_f16 = relu((A@B) + bias[n])
// EPI 2: out_f32 = (A@B) + bias[n]
template <int EPI>
__global__ __launch_bounds__(256) void gemm_f16_kernel(
    const f16* __restrict__ A, const f16* __restrict__ Bf, const float* __restrict__ bias,
    float* __restrict__ outF, f16* __restrict__ outH, int M, int Nn, int K) {
  __shared__ f16 As[2][128 * LDT];
  __shared__ f16 Bs[2][128 * LDT];
  __shared__ float rsum[128];
  int t = threadIdx.x, lane = t & 31, wid = t >> 5;
  int half = lane >> 4, nl = lane & 15;
  int wm = wid >> 2, wn = wid & 3;
  int m0 = blockIdx.y * 128, n0 = blockIdx.x * 128;
  v8f acc[4][2] = {};
  float rloc = 0.0f;
  const f16* Ab = A + (size_t)m0 * K;
  const f16* Bb = Bf + (size_t)n0 * K;
  int nk = K / 32;
  issue_tile(As[0], Ab, K, t);
  issue_tile(Bs[0], Bb, K, t);
  for (int kt = 0; kt < nk; ++kt) {
    int cur = kt & 1;
    wait_tile_loads();
    __syncthreads();
    if (kt + 1 < nk) {
      issue_tile(As[cur ^ 1], Ab + (kt + 1) * 32, K, t);
      issue_tile(Bs[cur ^ 1], Bb + (kt + 1) * 32, K, t);
    }
    if (EPI == 0) {  // per-row partial sums of A tile (2 threads per row)
      int rr = t >> 1, kh = (t & 1) << 4;
      float s = 0.0f;
#pragma unroll
      for (int k = 0; k < 16; ++k) s += (float)As[cur][rr * LDT + kh + k];
      rloc += s;
    }
    v16h af[4], bfr[2];
#pragma unroll
    for (int i = 0; i < 4; ++i) af[i] = frag_A(As[cur], (wm * 64 + i * 16 + nl) * LDT, half);
#pragma unroll
    for (int j = 0; j < 2; ++j) bfr[j] = frag_B(Bs[cur], (wn * 32 + j * 16 + nl) * LDT, half);
#pragma unroll
    for (int i = 0; i < 4; ++i)
#pragma unroll
      for (int j = 0; j < 2; ++j)
        acc[i][j] = __builtin_amdgcn_wmma_f32_16x16x32_f16(false, af[i], false, bfr[j],
                                                           (short)0, acc[i][j], false, false);
  }
  if (EPI == 0) {
    float tot = rloc + __shfl_xor(rloc, 1, 32);
    if ((t & 1) == 0) rsum[t >> 1] = tot;
    __syncthreads();
  }
#pragma unroll
  for (int i = 0; i < 4; ++i)
#pragma unroll
    for (int j = 0; j < 2; ++j)
#pragma unroll
      for (int r = 0; r < 8; ++r) {
        int row_l = wm * 64 + i * 16 + half * 8 + r;
        int col_l = wn * 32 + j * 16 + nl;
        size_t gi = (size_t)(m0 + row_l) * Nn + (n0 + col_l);
        float v = acc[i][j][r];
        if (EPI == 0) outF[gi] = v / rsum[row_l];
        if (EPI == 1) { v += bias[n0 + col_l]; outH[gi] = (f16)fmaxf(v, 0.0f); }
        if (EPI == 2) outF[gi] = v + bias[n0 + col_l];
      }
}

// ---------------- residual + LayerNorm (D == 1024, 256 thr, 4 elems/thr) -----------
__global__ __launch_bounds__(256) void add_ln_kernel(const float* __restrict__ X,
                                                     const float* __restrict__ Y,
                                                     const float* __restrict__ g,
                                                     const float* __restrict__ b,
                                                     float* __restrict__ outF,
                                                     f16* __restrict__ outH, int D) {
  __shared__ float red[8];
  __shared__ float stat;
  int row = blockIdx.x, t = threadIdx.x;
  const float* x = X + (size_t)row * D;
  const float* y = Y + (size_t)row * D;
  float v[4]; float s = 0.0f;
#pragma unroll
  for (int i = 0; i < 4; ++i) { int c = t + i * 256; float u = x[c] + y[c]; v[i] = u; s += u; }
#pragma unroll
  for (int m = 1; m < 32; m <<= 1) s += __shfl_xor(s, m, 32);
  if ((t & 31) == 0) red[t >> 5] = s;
  __syncthreads();
  if (t == 0) { float tt = 0; for (int k = 0; k < 8; ++k) tt += red[k]; stat = tt / (float)D; }
  __syncthreads();
  float mean = stat;
  float q = 0.0f;
#pragma unroll
  for (int i = 0; i < 4; ++i) { float d0 = v[i] - mean; q += d0 * d0; }
#pragma unroll
  for (int m = 1; m < 32; m <<= 1) q += __shfl_xor(q, m, 32);
  __syncthreads();
  if ((t & 31) == 0) red[t >> 5] = q;
  __syncthreads();
  if (t == 0) { float tt = 0; for (int k = 0; k < 8; ++k) tt += red[k];
                stat = rsqrtf(tt / (float)D + 1e-5f); }
  __syncthreads();
  float rstd = stat;
#pragma unroll
  for (int i = 0; i < 4; ++i) {
    int c = t + i * 256;
    float o = (v[i] - mean) * rstd * g[c] + b[c];
    outF[(size_t)row * D + c] = o;
    if (outH) outH[(size_t)row * D + c] = (f16)o;
  }
}

// ---------------- host orchestration ----------------------------------------------
extern "C" void kernel_launch(void* const* d_in, const int* in_sizes, int n_in,
                              void* d_out, int out_size, void* d_ws, size_t ws_size,
                              hipStream_t stream) {
  const float* src  = (const float*)d_in[0];
  const float* bbox = (const float*)d_in[1];
  const float* W1   = (const float*)d_in[2];
  const float* b1   = (const float*)d_in[3];
  const float* W2   = (const float*)d_in[4];
  const float* b2   = (const float*)d_in[5];
  const float* g1   = (const float*)d_in[6];
  const float* be1  = (const float*)d_in[7];
  const float* g2   = (const float*)d_in[8];
  const float* be2  = (const float*)d_in[9];
  float* out = (float*)d_out;

  const int N   = in_sizes[1] / 4;   // 8192
  const int D   = 1024;
  const int DFF = in_sizes[3];       // 4096

  char* ws = (char*)d_ws;
  size_t off = 0;
  auto alloc = [&](size_t bytes) { size_t o = off; off = (off + bytes + 255) & ~(size_t)255; return o; };
  float* cx    = (float*)(ws + alloc((size_t)N * 4));
  float* cy    = (float*)(ws + alloc((size_t)N * 4));
  float* scalF = (float*)(ws + alloc(256));            // [0]=dist_max [1]=bias_max
  unsigned int* scalU = (unsigned int*)scalF;
  f16* src16  = (f16*)(ws + alloc((size_t)N * D * 2));
  f16* srcT16 = (f16*)(ws + alloc((size_t)N * D * 2)); // [D][N]
  f16* W1t    = (f16*)(ws + alloc((size_t)DFF * D * 2)); // [DFF][D]
  f16* W2t    = (f16*)(ws + alloc((size_t)D * DFF * 2)); // [D][DFF]
  f16* P      = (f16*)(ws + alloc((size_t)N * N * 2));   // unnormalized probs
  float* attn = (float*)(ws + alloc((size_t)N * D * 4));
  float* h    = (float*)(ws + alloc((size_t)N * D * 4));
  f16* h16    = (f16*)(ws + alloc((size_t)N * D * 2));
  f16* Ff     = (f16*)(ws + alloc((size_t)N * DFF * 2));
  float* G    = (float*)(ws + alloc((size_t)N * D * 4));

  float scale = 1.0f / sqrtf((float)D);

  init_scal_kernel<<<1, 32, 0, stream>>>(scalU);
  centers_kernel<<<(N + 255) / 256, 256, 0, stream>>>(bbox, cx, cy, N);
  cast_f16_kernel<<<4096, 256, 0, stream>>>(src, src16, (size_t)N * D);
  transpose_cast_kernel<<<dim3(D / 32, N / 32), 256, 0, stream>>>(src, srcT16, N, D);
  transpose_cast_kernel<<<dim3(DFF / 32, D / 32), 256, 0, stream>>>(W1, W1t, D, DFF);
  transpose_cast_kernel<<<dim3(D / 32, DFF / 32), 256, 0, stream>>>(W2, W2t, DFF, D);
  pair_max_kernel<<<2048, 256, 0, stream>>>(cx, cy, scalF, scalU + 0, N, 0);
  pair_max_kernel<<<2048, 256, 0, stream>>>(cx, cy, scalF, scalU + 1, N, 1);

  attn_scores_kernel<<<dim3(N / 128, N / 128), 256, 0, stream>>>(src16, cx, cy, scalF, P,
                                                                 N, D, scale);
  gemm_f16_kernel<0><<<dim3(D / 128, N / 128), 256, 0, stream>>>(P, srcT16, nullptr,
                                                                 attn, nullptr, N, D, N);
  add_ln_kernel<<<N, 256, 0, stream>>>(attn, src, g1, be1, h, h16, D);
  gemm_f16_kernel<1><<<dim3(DFF / 128, N / 128), 256, 0, stream>>>(h16, W1t, b1,
                                                                   nullptr, Ff, N, DFF, D);
  gemm_f16_kernel<2><<<dim3(D / 128, N / 128), 256, 0, stream>>>(Ff, W2t, b2,
                                                                 G, nullptr, N, D, DFF);
  add_ln_kernel<<<N, 256, 0, stream>>>(h, G, g2, be2, out, nullptr, D);
}
